// MainNetwork_61916248539402
// MI455X (gfx1250) — matrix operand order repaired and checked by reading, not compiled, not run on hardware
//
#include <hip/hip_runtime.h>
#include <hip/hip_bf16.h>

// ---------------------------------------------------------------------------
// Types for WMMA fragments (gfx1250, wave32)
// ---------------------------------------------------------------------------
typedef _Float16 h8_t  __attribute__((ext_vector_type(8)));
typedef _Float16 h16_t __attribute__((ext_vector_type(16)));
typedef float    f8_t  __attribute__((ext_vector_type(8)));

__device__ __forceinline__ h16_t cat16(h8_t lo, h8_t hi) {
    h16_t r;
#pragma unroll
    for (int i = 0; i < 8; ++i) { r[i] = lo[i]; r[i + 8] = hi[i]; }
    return r;
}
__device__ __forceinline__ f8_t zero8() {
    f8_t z = {0.f, 0.f, 0.f, 0.f, 0.f, 0.f, 0.f, 0.f};
    return z;
}

// LDS byte-offset of a generic pointer that points into shared memory.
typedef __attribute__((address_space(3))) const void lds_cvoid_t;
__device__ __forceinline__ unsigned lds_off(const void* p) {
    return (unsigned)(uintptr_t)(lds_cvoid_t*)p;
}

// CDNA5 async global->LDS copy (16B per lane), tracked by ASYNCcnt.
__device__ __forceinline__ void async_copy_b128(unsigned dst_lds_off, const void* src) {
    asm volatile("global_load_async_to_lds_b128 %0, %1, off"
                 :
                 : "v"(dst_lds_off), "v"(src)
                 : "memory");
}
__device__ __forceinline__ void async_wait_all() {
    asm volatile("s_wait_asynccnt 0" ::: "memory");
}

// Geometry constants
#define NB   256   // batch
#define RR   32    // rooms
#define WW   8
#define HH   6
#define MXY  72
#define SPAT 5184  // 72*72
#define CIN1 16

// ---------------------------------------------------------------------------
// Kernel 1: convert all weight matrices to f16 in GEMM-friendly [o][K] layout.
// conv K order: K = tap*Cin + ci, tap = dx*kw + dy.  conv1 K padded 400 -> 416.
// ---------------------------------------------------------------------------
__global__ void prep_weights_kernel(
    const float* __restrict__ w1, const float* __restrict__ w2,
    const float* __restrict__ w3, const float* __restrict__ wr1,
    const float* __restrict__ wr2, const float* __restrict__ wf1,
    const float* __restrict__ wf2,
    _Float16* o1, _Float16* o2, _Float16* o3, _Float16* or1,
    _Float16* or2, _Float16* of1, _Float16* of2) {
    int i = blockIdx.x * 256 + threadIdx.x;

    if (i < 64 * 416) {  // conv1: [64][416] (400 real + 16 zero pad)
        int o = i / 416, k = i % 416;
        float v = 0.f;
        if (k < 400) {
            int tap = k / 16, ci = k % 16, dx = tap / 5, dy = tap % 5;
            v = w1[((o * 16 + ci) * 5 + dx) * 5 + dy];
        }
        o1[i] = (_Float16)v;
        return;
    }
    i -= 64 * 416;
    if (i < 64 * 576) {  // conv2
        int o = i / 576, k = i % 576;
        int tap = k / 64, ci = k % 64, dx = tap / 3, dy = tap % 3;
        o2[i] = (_Float16)w2[((o * 64 + ci) * 3 + dx) * 3 + dy];
        return;
    }
    i -= 64 * 576;
    if (i < 64 * 576) {  // conv3
        int o = i / 576, k = i % 576;
        int tap = k / 64, ci = k % 64, dx = tap / 3, dy = tap % 3;
        o3[i] = (_Float16)w3[((o * 64 + ci) * 3 + dx) * 3 + dy];
        return;
    }
    i -= 64 * 576;
    if (i < 128 * 64)  { or1[i] = (_Float16)wr1[i]; return; }
    i -= 128 * 64;
    if (i < 128 * 128) { or2[i] = (_Float16)wr2[i]; return; }
    i -= 128 * 128;
    if (i < 256 * 128) { of1[i] = (_Float16)wf1[i]; return; }
    i -= 256 * 128;
    if (i < 256 * 256) { of2[i] = (_Float16)wf2[i]; return; }
}

// ---------------------------------------------------------------------------
// Kernel 2: build the 16-channel input map directly as f16 [n][x][y][16].
// Gather-style (loop rooms per pixel) -> no atomics needed.
//   ch 0..8 : room_tensor scattered;  ch 9 : ones;  ch 10..15 : emb * room_map
// ---------------------------------------------------------------------------
__global__ void build_input_kernel(const int* __restrict__ pos,
                                   const float* __restrict__ rt,
                                   const float* __restrict__ emb,
                                   _Float16* __restrict__ Xh) {
    __shared__ int spx[RR];
    __shared__ int spy[RR];
    const int n = blockIdx.x / 21;
    const int s = (blockIdx.x % 21) * 256 + threadIdx.x;
    if (threadIdx.x < RR) {
        spx[threadIdx.x] = pos[(n * RR + threadIdx.x) * 2 + 0];
        spy[threadIdx.x] = pos[(n * RR + threadIdx.x) * 2 + 1];
    }
    __syncthreads();
    if (s >= SPAT) return;
    const int x = s / MXY, y = s % MXY;

    float acc[16];
#pragma unroll
    for (int c = 0; c < 16; ++c) acc[c] = 0.f;
    acc[9] = 1.f;

    for (int r = 0; r < RR; ++r) {
        int wx = x - spx[r];
        int hy = y - spy[r];
        if ((unsigned)wx < (unsigned)WW && (unsigned)hy < (unsigned)HH) {
            const float* base = rt + ((size_t)(r * 9) * WW + wx) * HH + hy;  // c stride 48
            float rm = base[0];
#pragma unroll
            for (int c = 0; c < 9; ++c) acc[c] += base[c * 48];
#pragma unroll
            for (int e = 0; e < 6; ++e) acc[10 + e] += emb[r * 6 + e] * rm;
        }
    }
    h8_t o0, o1;
#pragma unroll
    for (int c = 0; c < 8; ++c) { o0[c] = (_Float16)acc[c]; o1[c] = (_Float16)acc[c + 8]; }
    _Float16* dst = Xh + (size_t)(n * SPAT + s) * 16;
    *(h8_t*)dst       = o0;
    *(h8_t*)(dst + 8) = o1;
}

// ---------------------------------------------------------------------------
// Kernel 3: implicit-GEMM conv + bias + ReLU via v_wmma_f32_16x16x32_f16.
//   M = 64 out channels, N = 64 spatial positions per block, K = KLEN.
//   8 waves/block, each wave owns two 16x16 tiles.
//   Weights AND im2col tile staged in LDS via async global->LDS b128 copies,
//   so the steady-state K loop is pure ds_load_b128 + v_wmma.
//   Xin:  [n][x][y][CIN]   f16
//   Wf :  [64][KLEN]       f16   (K = tap*CIN + ci)
//   Yout: [n][x][y][64]    f16
// ---------------------------------------------------------------------------
template <int CIN, int KW, int KLEN>
__global__ void conv_wmma_kernel(const _Float16* __restrict__ Xin,
                                 const _Float16* __restrict__ Wf,
                                 const float* __restrict__ bias,
                                 _Float16* __restrict__ Yout) {
    constexpr int PAD = KW / 2;
    constexpr int T = KW * KW;
    constexpr int BSTRIDE = KLEN + 8;  // halves; keeps 16B alignment (mult of 8)
    extern __shared__ __align__(16) _Float16 smem[];
    _Float16* At = smem;               // [64][KLEN]   weights
    _Float16* Bt = smem + 64 * KLEN;   // [64][BSTRIDE] im2col rows

    const int tid = threadIdx.x;
    const int n = blockIdx.x / 81;
    const int s0 = (blockIdx.x % 81) * 64;

    // Zero the B tile (covers padding taps and conv1 K pad).
    {
        uint4 z; z.x = z.y = z.z = z.w = 0u;
        const int total16 = (64 * BSTRIDE) / 8;
        for (int i = tid; i < total16; i += 256) ((uint4*)Bt)[i] = z;
    }
    __syncthreads();

    // Async-stage the full weight tile [64][KLEN] (contiguous copy).
    {
        const int total16A = (64 * KLEN) / 8;
        for (int i = tid; i < total16A; i += 256)
            async_copy_b128(lds_off(At + i * 8), Wf + i * 8);
    }

    // Async-stage im2col rows: row = spatial position, contiguous K.
    const _Float16* Xn = Xin + (size_t)n * SPAT * CIN;
    for (int i = tid; i < 64 * T; i += 256) {
        const int row = i / T, tap = i % T;
        const int s = s0 + row;
        const int x = s / MXY, y = s % MXY;
        const int dx = tap / KW, dy = tap % KW;
        const int sx = x + dx - PAD, sy = y + dy - PAD;
        if ((unsigned)sx < (unsigned)MXY && (unsigned)sy < (unsigned)MXY) {
            const _Float16* src = Xn + (size_t)(sx * MXY + sy) * CIN;
            const unsigned dst = lds_off(Bt + row * BSTRIDE + tap * CIN);
#pragma unroll
            for (int q = 0; q < CIN / 8; ++q)
                async_copy_b128(dst + q * 16, src + q * 8);
        }
    }
    async_wait_all();   // each wave drains its own ASYNCcnt ...
    __syncthreads();    // ... then the workgroup barrier makes the tiles visible

    const int w = tid >> 5, lane = tid & 31;
    const int m15 = lane & 15;
    const int abase = (lane & 16) ? 8 : 0;    // A frag / C row base
    const int bbase = (lane & 16) ? 8 : 0;    // B frag K base

    for (int ti = 0; ti < 2; ++ti) {
        const int t = w + ti * 8;
        const int ot = t & 3;   // out-channel tile
        const int st = t >> 2;  // spatial sub-tile
        f8_t acc = zero8();
#pragma unroll
        for (int kc = 0; kc < KLEN / 32; ++kc) {
            const _Float16* ap = At + (ot * 16 + m15) * KLEN + kc * 32 + abase;
            h8_t alo = *(const h8_t*)ap;
            h8_t ahi = *(const h8_t*)(ap + 16);
            const _Float16* bp = Bt + (st * 16 + m15) * BSTRIDE + kc * 32 + bbase;
            h8_t blo = *(const h8_t*)bp;
            h8_t bhi = *(const h8_t*)(bp + 16);
            acc = __builtin_amdgcn_wmma_f32_16x16x32_f16(
                false, cat16(alo, ahi), false, cat16(blo, bhi),
                (short)0, acc, false, false);
        }
        // Epilogue: bias + ReLU, store 8 contiguous f16 channels.
        const int ncol = s0 + st * 16 + m15;       // spatial position
        const int obase = ot * 16 + abase;         // first of 8 channels
        h8_t outv;
#pragma unroll
        for (int r8 = 0; r8 < 8; ++r8) {
            float v = acc[r8] + bias[obase + r8];
            outv[r8] = (_Float16)(v > 0.f ? v : 0.f);
        }
        *(h8_t*)(Yout + ((size_t)n * SPAT + ncol) * 64 + obase) = outv;
    }
}

// ---------------------------------------------------------------------------
// Kernel 4: masked mean-pool of conv3 output over each room window.
//   feat[n][r][c] = sum_{w,h} Y[n][px+w][py+h][c]*mask / sum mask
// ---------------------------------------------------------------------------
__global__ void gather_pool_kernel(const _Float16* __restrict__ Y,
                                   const int* __restrict__ pos,
                                   const float* __restrict__ rt,
                                   _Float16* __restrict__ feat) {
    const int n = blockIdx.x / RR, r = blockIdx.x % RR;
    const int c = threadIdx.x;  // 64 threads
    const int px = pos[(n * RR + r) * 2 + 0];
    const int py = pos[(n * RR + r) * 2 + 1];
    float sum = 0.f, msum = 0.f;
    for (int wx = 0; wx < WW; ++wx)
        for (int hy = 0; hy < HH; ++hy) {
            float rm = rt[((size_t)(r * 9) * WW + wx) * HH + hy];
            float v = (float)Y[((size_t)n * SPAT + (size_t)(px + wx) * MXY + (py + hy)) * 64 + c];
            sum += v * rm;
            msum += rm;
        }
    feat[((size_t)n * RR + r) * 64 + c] = (_Float16)(sum / msum);
}

// ---------------------------------------------------------------------------
// Kernel 5: per-image room MLP 64->128->128 (ReLU) then sum over rooms.
//   GEMMs: M = out channels, N = 32 rooms, K = in channels.
// ---------------------------------------------------------------------------
__global__ void room_mlp_kernel(const _Float16* __restrict__ feat,
                                const _Float16* __restrict__ W1,
                                const float* __restrict__ b1,
                                const _Float16* __restrict__ W2,
                                const float* __restrict__ b2,
                                _Float16* __restrict__ S) {
    __shared__ __align__(16) _Float16 Btl[RR * 64];     // feat [r][c]
    __shared__ __align__(16) _Float16 inter[RR * 128];  // layer1 out [r][o]
    __shared__ __align__(16) float out2[RR * 128];      // layer2 out [r][o]
    const int n = blockIdx.x, tid = threadIdx.x;

    ((uint4*)Btl)[tid] = ((const uint4*)(feat + (size_t)n * RR * 64))[tid];  // 256*16B
    __syncthreads();

    const int w = tid >> 5, lane = tid & 31;
    const int m15 = lane & 15;
    const int abase = (lane & 16) ? 8 : 0;
    const int bbase = (lane & 16) ? 8 : 0;

    // Layer 1: M=128 (8 tiles) x N=32 (2 tiles), K=64
    for (int ti = 0; ti < 2; ++ti) {
        const int t = w + ti * 8;
        const int mt = t >> 1, nt = t & 1;
        f8_t acc = zero8();
#pragma unroll
        for (int kc = 0; kc < 2; ++kc) {
            const _Float16* ap = W1 + (size_t)(mt * 16 + m15) * 64 + kc * 32 + abase;
            h8_t alo = *(const h8_t*)ap, ahi = *(const h8_t*)(ap + 16);
            const _Float16* bp = Btl + (nt * 16 + m15) * 64 + kc * 32 + bbase;
            h8_t blo = *(const h8_t*)bp, bhi = *(const h8_t*)(bp + 16);
            acc = __builtin_amdgcn_wmma_f32_16x16x32_f16(
                false, cat16(alo, ahi), false, cat16(blo, bhi), (short)0, acc, false, false);
        }
        const int rcol = nt * 16 + m15;
        const int obase = mt * 16 + abase;
        h8_t o;
#pragma unroll
        for (int r8 = 0; r8 < 8; ++r8) {
            float v = acc[r8] + b1[obase + r8];
            o[r8] = (_Float16)(v > 0.f ? v : 0.f);
        }
        *(h8_t*)(inter + rcol * 128 + obase) = o;
    }
    __syncthreads();

    // Layer 2: M=128 x N=32, K=128
    for (int ti = 0; ti < 2; ++ti) {
        const int t = w + ti * 8;
        const int mt = t >> 1, nt = t & 1;
        f8_t acc = zero8();
#pragma unroll
        for (int kc = 0; kc < 4; ++kc) {
            const _Float16* ap = W2 + (size_t)(mt * 16 + m15) * 128 + kc * 32 + abase;
            h8_t alo = *(const h8_t*)ap, ahi = *(const h8_t*)(ap + 16);
            const _Float16* bp = inter + (nt * 16 + m15) * 128 + kc * 32 + bbase;
            h8_t blo = *(const h8_t*)bp, bhi = *(const h8_t*)(bp + 16);
            acc = __builtin_amdgcn_wmma_f32_16x16x32_f16(
                false, cat16(alo, ahi), false, cat16(blo, bhi), (short)0, acc, false, false);
        }
        const int rcol = nt * 16 + m15;
        const int obase = mt * 16 + abase;
#pragma unroll
        for (int r8 = 0; r8 < 8; ++r8) {
            float v = acc[r8] + b2[obase + r8];
            out2[rcol * 128 + obase + r8] = v > 0.f ? v : 0.f;
        }
    }
    __syncthreads();

    // Sum over rooms -> S[n][128] f16
    if (tid < 128) {
        float s = 0.f;
        for (int r = 0; r < RR; ++r) s += out2[r * 128 + tid];
        S[(size_t)n * 128 + tid] = (_Float16)s;
    }
}

// ---------------------------------------------------------------------------
// Kernel 6: FC head. 16 images/block: y1 = relu(Wfc1 @ s + b1); out = Wfc2 @ y1 + b2
//   GEMMs: M = 256 outputs (16 tiles), N = 16 images (1 tile), K = 128 / 256.
// ---------------------------------------------------------------------------
__global__ void fc_kernel(const _Float16* __restrict__ S,
                          const _Float16* __restrict__ W1,
                          const float* __restrict__ b1,
                          const _Float16* __restrict__ W2,
                          const float* __restrict__ b2,
                          float* __restrict__ out) {
    __shared__ __align__(16) _Float16 Btl[16 * 128];  // S rows
    __shared__ __align__(16) _Float16 Y1[16 * 256];   // hidden
    const int n0 = blockIdx.x * 16, tid = threadIdx.x;

    ((uint4*)Btl)[tid] = ((const uint4*)(S + (size_t)n0 * 128))[tid];  // 256*16B
    __syncthreads();

    const int w = tid >> 5, lane = tid & 31;
    const int m15 = lane & 15;  // image column within block for B/C
    const int abase = (lane & 16) ? 8 : 0;
    const int bbase = (lane & 16) ? 8 : 0;

    // fc1: K=128
    for (int ti = 0; ti < 2; ++ti) {
        const int mt = w + ti * 8;
        f8_t acc = zero8();
#pragma unroll
        for (int kc = 0; kc < 4; ++kc) {
            const _Float16* ap = W1 + (size_t)(mt * 16 + m15) * 128 + kc * 32 + abase;
            h8_t alo = *(const h8_t*)ap, ahi = *(const h8_t*)(ap + 16);
            const _Float16* bp = Btl + m15 * 128 + kc * 32 + bbase;
            h8_t blo = *(const h8_t*)bp, bhi = *(const h8_t*)(bp + 16);
            acc = __builtin_amdgcn_wmma_f32_16x16x32_f16(
                false, cat16(alo, ahi), false, cat16(blo, bhi), (short)0, acc, false, false);
        }
        const int obase = mt * 16 + abase;
        h8_t o;
#pragma unroll
        for (int r8 = 0; r8 < 8; ++r8) {
            float v = acc[r8] + b1[obase + r8];
            o[r8] = (_Float16)(v > 0.f ? v : 0.f);
        }
        *(h8_t*)(Y1 + m15 * 256 + obase) = o;
    }
    __syncthreads();

    // fc2: K=256, fp32 output
    for (int ti = 0; ti < 2; ++ti) {
        const int mt = w + ti * 8;
        f8_t acc = zero8();
#pragma unroll
        for (int kc = 0; kc < 8; ++kc) {
            const _Float16* ap = W2 + (size_t)(mt * 16 + m15) * 256 + kc * 32 + abase;
            h8_t alo = *(const h8_t*)ap, ahi = *(const h8_t*)(ap + 16);
            const _Float16* bp = Y1 + m15 * 256 + kc * 32 + bbase;
            h8_t blo = *(const h8_t*)bp, bhi = *(const h8_t*)(bp + 16);
            acc = __builtin_amdgcn_wmma_f32_16x16x32_f16(
                false, cat16(alo, ahi), false, cat16(blo, bhi), (short)0, acc, false, false);
        }
        const int obase = mt * 16 + abase;
#pragma unroll
        for (int r8 = 0; r8 < 8; ++r8)
            out[(size_t)(n0 + m15) * 256 + obase + r8] = acc[r8] + b2[obase + r8];
    }
}

// ---------------------------------------------------------------------------
// Host launcher
// ---------------------------------------------------------------------------
extern "C" void kernel_launch(void* const* d_in, const int* in_sizes, int n_in,
                              void* d_out, int out_size, void* d_ws, size_t ws_size,
                              hipStream_t stream) {
    (void)in_sizes; (void)n_in; (void)out_size; (void)ws_size;
    const int*   pos = (const int*)d_in[0];
    const float* rt  = (const float*)d_in[1];
    const float* emb = (const float*)d_in[2];
    const float* w1  = (const float*)d_in[3];
    const float* b1  = (const float*)d_in[4];
    const float* w2  = (const float*)d_in[5];
    const float* b2  = (const float*)d_in[6];
    const float* w3  = (const float*)d_in[7];
    const float* b3  = (const float*)d_in[8];
    const float* wr1 = (const float*)d_in[9];
    const float* br1 = (const float*)d_in[10];
    const float* wr2 = (const float*)d_in[11];
    const float* br2 = (const float*)d_in[12];
    const float* wf1 = (const float*)d_in[13];
    const float* bf1 = (const float*)d_in[14];
    const float* wf2 = (const float*)d_in[15];
    const float* bf2 = (const float*)d_in[16];

    char* p = (char*)d_ws;
    auto alloc = [&](size_t bytes) -> char* {
        char* q = p;
        p += (bytes + 255) & ~(size_t)255;
        return q;
    };
    _Float16* w1h  = (_Float16*)alloc((size_t)64 * 416 * 2);
    _Float16* w2h  = (_Float16*)alloc((size_t)64 * 576 * 2);
    _Float16* w3h  = (_Float16*)alloc((size_t)64 * 576 * 2);
    _Float16* wr1h = (_Float16*)alloc((size_t)128 * 64 * 2);
    _Float16* wr2h = (_Float16*)alloc((size_t)128 * 128 * 2);
    _Float16* wf1h = (_Float16*)alloc((size_t)256 * 128 * 2);
    _Float16* wf2h = (_Float16*)alloc((size_t)256 * 256 * 2);
    _Float16* Xh   = (_Float16*)alloc((size_t)NB * SPAT * 16 * 2);
    _Float16* A1   = (_Float16*)alloc((size_t)NB * SPAT * 64 * 2);
    _Float16* A2   = (_Float16*)alloc((size_t)NB * SPAT * 64 * 2);
    _Float16* feat = (_Float16*)alloc((size_t)NB * RR * 64 * 2);
    _Float16* Sh   = (_Float16*)alloc((size_t)NB * 128 * 2);

    prep_weights_kernel<<<872, 256, 0, stream>>>(w1, w2, w3, wr1, wr2, wf1, wf2,
                                                 w1h, w2h, w3h, wr1h, wr2h, wf1h, wf2h);
    build_input_kernel<<<NB * 21, 256, 0, stream>>>(pos, rt, emb, Xh);

    conv_wmma_kernel<16, 5, 416>
        <<<NB * 81, 256, (size_t)(64 * 416 + 64 * (416 + 8)) * 2, stream>>>(Xh, w1h, b1, A1);
    conv_wmma_kernel<64, 3, 576>
        <<<NB * 81, 256, (size_t)(64 * 576 + 64 * (576 + 8)) * 2, stream>>>(A1, w2h, b2, A2);
    conv_wmma_kernel<64, 3, 576>
        <<<NB * 81, 256, (size_t)(64 * 576 + 64 * (576 + 8)) * 2, stream>>>(A2, w3h, b3, A1);

    gather_pool_kernel<<<NB * RR, 64, 0, stream>>>(A1, pos, rt, feat);
    room_mlp_kernel<<<NB, 256, 0, stream>>>(feat, wr1h, br1, wr2h, br2, Sh);
    fc_kernel<<<16, 256, 0, stream>>>(Sh, wf1h, bf1, wf2h, bf2, (float*)d_out);
}
